// MultiHead_Attn_40913858461857
// MI455X (gfx1250) — compile-verified, matrix operand order checked by reading, not run on hardware
//
#include <hip/hip_runtime.h>
#include <hip/hip_bf16.h>
#include <math.h>

// ---------------------------------------------------------------------------
// MultiHead attention block for MI455X (gfx1250), wave32 + WMMA f16->f32.
// B=256, S=127, D=1024, H=16, HD=64.  M = B*S = 32512 (divisible by 64).
// B operands staged N-major ([n][k]) so every WMMA fragment is two LDS b128
// loads; GEMM global->LDS staging uses gfx1250 async direct-to-LDS copies.
// ---------------------------------------------------------------------------

typedef _Float16 half_t;
typedef __attribute__((ext_vector_type(16))) _Float16 v16h;
typedef __attribute__((ext_vector_type(8)))  _Float16 v8h;
typedef __attribute__((ext_vector_type(4)))  _Float16 v4h;
typedef __attribute__((ext_vector_type(8)))  float    v8f;

#define BB   256
#define SS   127
#define SPAD 128
#define DD   1024
#define HH   16
#define HDIM 64

// ---- k-major fragment loader (used for A and for N-major-staged B) --------
// Lane L: row = L&15; lanes<16 hold K 0..7 & 16..23, lanes>=16 hold
// K 8..15 & 24..31  -> two contiguous 8-half (16B) LDS loads per lane.
__device__ inline v16h load_frag_km(const half_t* __restrict__ src, int ld) {
  int lane = threadIdx.x & 31;
  int r  = lane & 15;
  int kb = (lane & 16) >> 1;           // 0 or 8
  const half_t* p = src + r * ld + kb;
  v8h lo = *(const v8h*)(p);
  v8h hi = *(const v8h*)(p + 16);
  return __builtin_shufflevector(lo, hi, 0, 1, 2, 3, 4, 5, 6, 7,
                                         8, 9, 10, 11, 12, 13, 14, 15);
}

__device__ inline v8f wmma_f16(v16h a, v16h b, v8f c) {
  return __builtin_amdgcn_wmma_f32_16x16x32_f16(false, a, false, b,
                                                (short)0, c, false, false);
}

// ---- gfx1250 async global->LDS copy (ASYNCcnt-tracked) --------------------
// Flat shared address low 32 bits == LDS byte offset (aperture layout).
__device__ inline unsigned lds_off_of(const void* p) {
  return (unsigned)(unsigned long long)p;
}
__device__ inline void async_ld_b128(unsigned lds_off, const half_t* g) {
  asm volatile("global_load_async_to_lds_b128 %0, %1, off"
               :: "v"(lds_off), "v"(g) : "memory");
}
__device__ inline void wait_asynccnt0() {
  asm volatile("s_wait_asynccnt 0" ::: "memory");
}

// ---------------- LayerNorm fp32 -> fp16 (one block per row, 1024 cols) ----
__global__ __launch_bounds__(256)
void layernorm_f16_kernel(const float* __restrict__ x,
                          const float* __restrict__ gamma,
                          const float* __restrict__ beta,
                          half_t* __restrict__ y) {
  int row = blockIdx.x;
  int tid = threadIdx.x;
  const float* xr = x + (size_t)row * DD;
  float4 v = *(const float4*)&xr[tid * 4];
  float s  = v.x + v.y + v.z + v.w;
  float s2 = v.x * v.x + v.y * v.y + v.z * v.z + v.w * v.w;

  __shared__ float rs[256];
  __shared__ float rs2[256];
  rs[tid] = s; rs2[tid] = s2;
  __syncthreads();
  for (int off = 128; off > 0; off >>= 1) {
    if (tid < off) { rs[tid] += rs[tid + off]; rs2[tid] += rs2[tid + off]; }
    __syncthreads();
  }
  float mean = rs[0] * (1.0f / DD);
  float var  = rs2[0] * (1.0f / DD) - mean * mean;
  float rstd = rsqrtf(fmaxf(var, 0.0f) + 1e-5f);

  half_t* yr = y + (size_t)row * DD;
  float gv[4] = {gamma[tid*4], gamma[tid*4+1], gamma[tid*4+2], gamma[tid*4+3]};
  float bv[4] = {beta[tid*4],  beta[tid*4+1],  beta[tid*4+2],  beta[tid*4+3]};
  float xv[4] = {v.x, v.y, v.z, v.w};
#pragma unroll
  for (int i = 0; i < 4; ++i)
    yr[tid*4 + i] = (half_t)((xv[i] - mean) * rstd * gv[i] + bv[i]);
}

// ---------------- fp32 -> fp16 cast (weights kept in native [n][k]) --------
__global__ __launch_bounds__(256)
void cast_f16_kernel(const float* __restrict__ W, half_t* __restrict__ Wh) {
  int idx = (blockIdx.x * 256 + threadIdx.x) * 4;
  float4 v = *(const float4*)&W[idx];
  v4h o;
  o[0] = (half_t)v.x; o[1] = (half_t)v.y;
  o[2] = (half_t)v.z; o[3] = (half_t)v.w;
  *(v4h*)&Wh[idx] = o;
}

// ---------------- WMMA GEMM: C[MxN] = A[MxK] * B^T, B given N-major [N][K] -
// Block tile 64x128, 128 threads = 4 waves; wave w -> rows 16w..16w+15, all
// 128 cols (8 accumulators). K consumed 32/iteration (one WMMA k-step).
#define ALD 40    // 32 + 8 halfs pad (80B rows)
#define BLD 40
__global__ __launch_bounds__(128)
void gemm_wmma_kernel(const half_t* __restrict__ A,     // [M][K]
                      const half_t* __restrict__ Bm,    // [N][K] (= torch W)
                      const float*  __restrict__ bias,
                      const float*  __restrict__ resid, // null unless f32 out
                      half_t* __restrict__ outH,
                      float*  __restrict__ outF,
                      int M, int N, int K) {
  __shared__ half_t As[64 * ALD];    // 64 rows (m) x 32 (k)
  __shared__ half_t Bs[128 * BLD];   // 128 rows (n) x 32 (k)
  int tid  = threadIdx.x;
  int wave = tid >> 5;
  int lane = tid & 31;
  int m0 = blockIdx.y * 64;
  int n0 = blockIdx.x * 128;

  // fixed per-thread staging chunks (8 halfs = 16B each)
  const half_t* pA[2]; unsigned lA[2];
#pragma unroll
  for (int j = 0; j < 2; ++j) {
    int c = tid + j * 128;             // 256 chunks: 64 rows x 4
    int r = c >> 2, cc = (c & 3) << 3;
    pA[j] = A + (size_t)(m0 + r) * K + cc;
    lA[j] = lds_off_of(&As[r * ALD + cc]);
  }
  const half_t* pB[4]; unsigned lB[4];
#pragma unroll
  for (int j = 0; j < 4; ++j) {
    int c = tid + j * 128;             // 512 chunks: 128 rows x 4
    int r = c >> 2, cc = (c & 3) << 3;
    pB[j] = Bm + (size_t)(n0 + r) * K + cc;
    lB[j] = lds_off_of(&Bs[r * BLD + cc]);
  }

  v8f acc[8];
#pragma unroll
  for (int i = 0; i < 8; ++i) { v8f z = {}; acc[i] = z; }

  for (int k0 = 0; k0 < K; k0 += 32) {
#pragma unroll
    for (int j = 0; j < 2; ++j) async_ld_b128(lA[j], pA[j] + k0);
#pragma unroll
    for (int j = 0; j < 4; ++j) async_ld_b128(lB[j], pB[j] + k0);
    wait_asynccnt0();                  // own wave's LDS writes landed
    __syncthreads();                   // everyone's landed

    v16h af = load_frag_km(&As[(wave * 16) * ALD], ALD);
#pragma unroll
    for (int nt = 0; nt < 8; ++nt) {
      v16h bf = load_frag_km(&Bs[(nt * 16) * BLD], BLD);
      acc[nt] = wmma_f16(af, bf, acc[nt]);
    }
    __syncthreads();
  }

  int rbase = (lane & 16) ? 8 : 0;
  int cn    = lane & 15;
#pragma unroll
  for (int nt = 0; nt < 8; ++nt) {
#pragma unroll
    for (int r = 0; r < 8; ++r) {
      int gm = m0 + wave * 16 + r + rbase;
      int gn = n0 + nt * 16 + cn;
      if (gm < M) {
        float v = acc[nt][r] + bias[gn];
        if (outF) outF[(size_t)gm * N + gn] = v + resid[(size_t)gm * N + gn];
        else      outH[(size_t)gm * N + gn] = (half_t)v;
      }
    }
  }
}

// ---------------- Attention: per (b, h, 16-row q tile) ---------------------
// scores = (Qt Kt^T)/8, mask -> -inf, softmax, O = P V. GEMMs via WMMA.
#define QLD 72    // 64 + 8 halfs (144B rows)
#define KLD 72
#define VLD 136   // 128 + 8 halfs (272B rows)
#define PLD 136
__global__ __launch_bounds__(128)
void attn_kernel(const half_t* __restrict__ qb,
                 const half_t* __restrict__ kb,
                 const half_t* __restrict__ vb,
                 const unsigned char* __restrict__ mask,   // [B,1,S,S] bool
                 half_t* __restrict__ ob) {
  int qt = blockIdx.x, h = blockIdx.y, b = blockIdx.z;
  int tid  = threadIdx.x;
  int lane = tid & 31;
  int wave = tid >> 5;
  int q0 = qt * 16;
  int colbase = h * HDIM;

  __shared__ half_t Qs[16 * QLD];         // [m][d]
  __shared__ half_t Ks[SPAD * KLD];       // [kk][d]  (B for scores, N-major)
  __shared__ half_t VsT[HDIM * VLD];      // [d][kk]  (B for PV, N-major)
  __shared__ float  Sc[16 * SPAD];
  __shared__ half_t Ps[16 * PLD];
  __shared__ float  red[16 * 8];
  __shared__ float  rowinv[16];

  // Q tile 16x64 (pad rows -> 0); 128 chunks of 8 halfs, one per thread
  {
    int r = tid >> 3, c = (tid & 7) << 3;
    int qrow = q0 + r;
    if (qrow < SS) {
      *(v8h*)&Qs[r * QLD + c] =
          *(const v8h*)&qb[((size_t)b * SS + qrow) * DD + colbase + c];
    } else {
      v8h z = {};
      *(v8h*)&Qs[r * QLD + c] = z;
    }
  }
  // K tile 128x64 row-major (coalesced, zero pad row 127)
  for (int c = tid; c < SPAD * 8; c += 128) {
    int kk = c >> 3, d = (c & 7) << 3;
    if (kk < SS) {
      *(v8h*)&Ks[kk * KLD + d] =
          *(const v8h*)&kb[((size_t)b * SS + kk) * DD + colbase + d];
    } else {
      v8h z = {};
      *(v8h*)&Ks[kk * KLD + d] = z;
    }
  }
  // V^T 64x128 (coalesced read, transposed LDS write; pad col 127 -> 0)
  for (int idx = tid; idx < SPAD * HDIM; idx += 128) {
    int kk = idx >> 6, d = idx & 63;
    half_t v = (half_t)0.0f;
    if (kk < SS) v = vb[((size_t)b * SS + kk) * DD + colbase + d];
    VsT[d * VLD + kk] = v;
  }
  __syncthreads();

  // ---- scores: 16x128; wave w owns n-tiles {2w, 2w+1}; K=64 in 2 steps ----
  {
    v8f sacc0 = {}, sacc1 = {};
#pragma unroll
    for (int ks = 0; ks < 2; ++ks) {
      v16h af = load_frag_km(&Qs[ks * 32], QLD);
      v16h b0 = load_frag_km(&Ks[((wave * 2 + 0) * 16) * KLD + ks * 32], KLD);
      sacc0 = wmma_f16(af, b0, sacc0);
      v16h b1 = load_frag_km(&Ks[((wave * 2 + 1) * 16) * KLD + ks * 32], KLD);
      sacc1 = wmma_f16(af, b1, sacc1);
    }
    int rbase = (lane & 16) ? 8 : 0;
    int cn    = lane & 15;
#pragma unroll
    for (int r = 0; r < 8; ++r) {
      Sc[(r + rbase) * SPAD + (wave * 2 + 0) * 16 + cn] = sacc0[r];
      Sc[(r + rbase) * SPAD + (wave * 2 + 1) * 16 + cn] = sacc1[r];
    }
  }
  __syncthreads();

  // ---- scale + mask ----
  const float NEG_INF = -__builtin_inff();
  for (int idx = tid; idx < 16 * SPAD; idx += 128) {
    int row = idx >> 7, col = idx & 127;
    int qrow = q0 + row;
    float v = NEG_INF;
    if (qrow < SS && col < SS) {
      unsigned char mk = mask[((size_t)b * SS + qrow) * SS + col];
      v = mk ? NEG_INF : Sc[idx] * 0.125f;
    }
    Sc[idx] = v;
  }
  __syncthreads();

  // ---- softmax (8 threads/row); P kept unnormalized, 1/sum folded later ---
  {
    int srow = tid >> 3, sj = tid & 7;
    float pmax = NEG_INF;
    for (int col = sj; col < SPAD; col += 8)
      pmax = fmaxf(pmax, Sc[srow * SPAD + col]);
    red[srow * 8 + sj] = pmax;
    __syncthreads();
    float rmax = NEG_INF;
#pragma unroll
    for (int t = 0; t < 8; ++t) rmax = fmaxf(rmax, red[srow * 8 + t]);
    __syncthreads();
    float psum = 0.0f;
    for (int col = sj; col < SPAD; col += 8) {
      float sv = Sc[srow * SPAD + col];
      float p = (rmax == NEG_INF || sv == NEG_INF) ? 0.0f : __expf(sv - rmax);
      Ps[srow * PLD + col] = (half_t)p;
      psum += p;
    }
    red[srow * 8 + sj] = psum;
    __syncthreads();
    if (sj == 0) {
      float rsum = 0.0f;
#pragma unroll
      for (int t = 0; t < 8; ++t) rsum += red[srow * 8 + t];
      rowinv[srow] = (rsum > 0.0f) ? 1.0f / rsum : 0.0f;
    }
  }
  __syncthreads();

  // ---- O = P(16x128) * V(128x64); wave w owns output cols 16w..16w+15 ----
  {
    v8f oacc = {};
#pragma unroll
    for (int ks = 0; ks < 4; ++ks) {
      v16h af = load_frag_km(&Ps[ks * 32], PLD);
      v16h bf = load_frag_km(&VsT[(wave * 16) * VLD + ks * 32], VLD);
      oacc = wmma_f16(af, bf, oacc);
    }
    int rbase = (lane & 16) ? 8 : 0;
    int cn    = lane & 15;
#pragma unroll
    for (int r = 0; r < 8; ++r) {
      int row  = r + rbase;
      int qrow = q0 + row;
      if (qrow < SS) {
        ob[((size_t)b * SS + qrow) * DD + colbase + wave * 16 + cn] =
            (half_t)(oacc[r] * rowinv[row]);
      }
    }
  }
}

// ---------------------------------------------------------------------------
extern "C" void kernel_launch(void* const* d_in, const int* in_sizes, int n_in,
                              void* d_out, int out_size, void* d_ws, size_t ws_size,
                              hipStream_t stream) {
  const float* Q     = (const float*)d_in[0];
  const float* K     = (const float*)d_in[1];
  const float* V     = (const float*)d_in[2];
  const unsigned char* mask = (const unsigned char*)d_in[3];  // jnp bool = 1B
  const float* Wq = (const float*)d_in[4];
  const float* bq = (const float*)d_in[5];
  const float* Wk = (const float*)d_in[6];
  const float* bk = (const float*)d_in[7];
  const float* Wv = (const float*)d_in[8];
  const float* bv = (const float*)d_in[9];
  const float* Wo = (const float*)d_in[10];
  const float* bo = (const float*)d_in[11];
  const float* gamma = (const float*)d_in[12];
  const float* beta  = (const float*)d_in[13];

  const size_t M = (size_t)BB * SS;               // 32512 = 508 * 64
  char* ws = (char*)d_ws;
  size_t off = 0;
  half_t* Xn  = (half_t*)(ws + off); off += M * DD * sizeof(half_t); // + attn out
  half_t* Wqh = (half_t*)(ws + off); off += (size_t)DD * DD * sizeof(half_t);
  half_t* Wkh = (half_t*)(ws + off); off += (size_t)DD * DD * sizeof(half_t);
  half_t* Wvh = (half_t*)(ws + off); off += (size_t)DD * DD * sizeof(half_t);
  half_t* Woh = (half_t*)(ws + off); off += (size_t)DD * DD * sizeof(half_t);
  half_t* qbuf = (half_t*)(ws + off); off += M * DD * sizeof(half_t);
  half_t* kbuf = (half_t*)(ws + off); off += M * DD * sizeof(half_t);
  half_t* vbuf = (half_t*)(ws + off); off += M * DD * sizeof(half_t);

  const unsigned castGrid = (DD * DD) / (256 * 4);  // 1024
  cast_f16_kernel<<<castGrid, 256, 0, stream>>>(Wq, Wqh);
  cast_f16_kernel<<<castGrid, 256, 0, stream>>>(Wk, Wkh);
  cast_f16_kernel<<<castGrid, 256, 0, stream>>>(Wv, Wvh);
  cast_f16_kernel<<<castGrid, 256, 0, stream>>>(Wo, Woh);

  dim3 ggrid(DD / 128, (unsigned)(M / 64));       // (8, 508)

  // q = LN(Q) @ Wq^T + bq
  layernorm_f16_kernel<<<(unsigned)M, 256, 0, stream>>>(Q, gamma, beta, Xn);
  gemm_wmma_kernel<<<ggrid, 128, 0, stream>>>(Xn, Wqh, bq, nullptr,
                                              qbuf, nullptr, (int)M, DD, DD);
  // k = LN(K) @ Wk^T + bk
  layernorm_f16_kernel<<<(unsigned)M, 256, 0, stream>>>(K, gamma, beta, Xn);
  gemm_wmma_kernel<<<ggrid, 128, 0, stream>>>(Xn, Wkh, bk, nullptr,
                                              kbuf, nullptr, (int)M, DD, DD);
  // v = LN(V) @ Wv^T + bv
  layernorm_f16_kernel<<<(unsigned)M, 256, 0, stream>>>(V, gamma, beta, Xn);
  gemm_wmma_kernel<<<ggrid, 128, 0, stream>>>(Xn, Wvh, bv, nullptr,
                                              vbuf, nullptr, (int)M, DD, DD);

  // attention -> ob (reuse Xn; LN values are dead now)
  dim3 agrid((SS + 15) / 16, HH, BB);             // (8, 16, 256)
  attn_kernel<<<agrid, 128, 0, stream>>>(qbuf, kbuf, vbuf, mask, Xn);

  // out = resid + ob @ Wo^T + bo   (fp32)
  gemm_wmma_kernel<<<ggrid, 128, 0, stream>>>(Xn, Woh, bo, Q,
                                              nullptr, (float*)d_out,
                                              (int)M, DD, DD);
}
